// FNO1DAdvanced_8847632629995
// MI455X (gfx1250) — matrix operand order repaired and checked by reading, not compiled, not run on hardware
//
#include <hip/hip_runtime.h>
#include <math.h>

typedef __bf16 bf16;
typedef __attribute__((ext_vector_type(16))) __bf16 v16bf;
typedef __attribute__((ext_vector_type(8)))  float  v8f;

#define BB   32
#define NN   8192
#define CC   128
#define MODES2 128   // 64 modes * (re,im)
#define NLAYERS 6

__device__ __forceinline__ float gelu_f(float x) {
    return 0.5f * x * (1.0f + erff(x * 0.70710678118654752f));
}

// ---- WMMA helpers (16x16x32 bf16, f32 accum) -------------------------------
// A (16x32): lane l holds row m=l%16; element e -> K = e + 16*half
// B (32x16): lane l holds col n=l%16; element e -> K = e + 16*half
// C (16x16): lane l col n=l%16; vgpr v -> row m = v + 8*half

__device__ __forceinline__ v8f wmma_bf16(v16bf a, v16bf b, v8f c) {
    return __builtin_amdgcn_wmma_f32_16x16x32_bf16(false, a, false, b, (short)0, c,
                                                   false, false);
}

// A fragment from bf16, K-contiguous row (two 16B runs)
__device__ __forceinline__ v16bf load_fragA_bf16(const bf16* row, int kbase) {
    int lane = threadIdx.x & 31;
    int half = lane >> 4;
    const bf16* p0 = row + kbase + 16 * half;        // e = 0..7
    const bf16* p1 = p0 + 8;                         // e = 8..15
    v16bf a;
#pragma unroll
    for (int e = 0; e < 8; ++e) { a[e] = p0[e]; a[8 + e] = p1[e]; }
    return a;
}

// A fragment from f32 (converted), K-contiguous row
__device__ __forceinline__ v16bf load_fragA_f32(const float* row, int kbase) {
    int lane = threadIdx.x & 31;
    int half = lane >> 4;
    const float* p0 = row + kbase + 16 * half;
    const float* p1 = p0 + 8;
    v16bf a;
#pragma unroll
    for (int e = 0; e < 8; ++e) { a[e] = (bf16)p0[e]; a[8 + e] = (bf16)p1[e]; }
    return a;
}

// B fragment from bf16 stored (col, K) K-contiguous (one 32B run per lane)
__device__ __forceinline__ v16bf load_fragB_bf16(const bf16* row, int kbase) {
    int lane = threadIdx.x & 31;
    int half = lane >> 4;
    const bf16* p = row + kbase + 16 * half;
    v16bf b;
#pragma unroll
    for (int e = 0; e < 16; ++e) b[e] = p[e];
    return b;
}

__device__ __forceinline__ void store_tile(float* C, int ldc, int m0, int n0, v8f acc) {
    int lane = threadIdx.x & 31;
    int half = lane >> 4, nc = lane & 15;
#pragma unroll
    for (int v = 0; v < 8; ++v)
        C[(size_t)(m0 + v + 8 * half) * ldc + n0 + nc] = acc[v];
}

// ---- DFT matrix init -------------------------------------------------------
__global__ void init_dft_kernel(bf16* Ft, bf16* Gt) {
    int idx = blockIdx.x * blockDim.x + threadIdx.x;   // 128*8192 entries
    int n  = idx & (NN - 1);
    int m2 = idx >> 13;
    int m  = m2 >> 1, r = m2 & 1;
    int ph = (m * n) & (NN - 1);                        // exact phase reduction
    float th = 6.2831853071795864f * (float)ph / (float)NN;
    float c = cosf(th), s = sinf(th);
    float fv = r ? -s : c;
    Ft[(size_t)m2 * NN + n] = (bf16)fv;
    float scale = (m == 0) ? (1.0f / (float)NN) : (2.0f / (float)NN);
    float gv = r ? (-s * scale) : (c * scale);
    if (m == 0 && r == 1) gv = 0.0f;                    // irfft ignores imag(DC)
    Gt[(size_t)n * MODES2 + m2] = (bf16)gv;
}

// ---- lifting: conv1x1(1->64) -> gelu -> conv1x1(64->128) -------------------
__global__ void lift_kernel(const float* __restrict__ src,
                            const float* __restrict__ w1, const float* __restrict__ b1,
                            const float* __restrict__ w2, const float* __restrict__ b2,
                            float* __restrict__ X, bf16* __restrict__ Xb) {
    int idx = blockIdx.x * blockDim.x + threadIdx.x;   // B*N
    int n = idx & (NN - 1);
    int b = idx >> 13;
    float s = src[idx];
    float h[64];
#pragma unroll
    for (int j = 0; j < 64; ++j) h[j] = gelu_f(w1[j] * s + b1[j]);
    float* Xp = X + (size_t)b * CC * NN + n;
    bf16*  Bp = Xb + (size_t)b * CC * NN + n;
    for (int c = 0; c < CC; ++c) {
        float acc = b2[c];
        const float* w = w2 + c * 64;
#pragma unroll 8
        for (int j = 0; j < 64; ++j) acc += w[j] * h[j];
        Xp[(size_t)c * NN] = acc;
        Bp[(size_t)c * NN] = (bf16)acc;
    }
}

// ---- forward DFT GEMM: Xhat(b)[c][m2] = Xb(b)[c][n] * Ft[m2][n]^T ----------
// Each wave: 16 c-rows x 64 m2-cols. Ping-pong double buffer; launch_bounds
// (.,1) lifts the VGPR budget so both fragment sets + accumulators stay in
// registers (no scratch spills).
__global__ void __launch_bounds__(128, 1)
dft_fwd_kernel(const bf16* __restrict__ Xb,
               const bf16* __restrict__ Ft,
               float* __restrict__ Xhat) {
    // grid (2, 2, B), block (32,4): tm = by*4+ty (8 c-tiles), n0 = bx*64
    int tm = blockIdx.y * 4 + threadIdx.y;
    int n0 = blockIdx.x * 64;
    int b  = blockIdx.z;
    const bf16* Arow = Xb + ((size_t)b * CC + tm * 16 + (threadIdx.x & 15)) * NN;
    const bf16* B0 = Ft + (size_t)(n0 + (threadIdx.x & 15)) * NN;
    const bf16* B1 = B0 + (size_t)16 * NN;
    const bf16* B2 = B0 + (size_t)32 * NN;
    const bf16* B3 = B0 + (size_t)48 * NN;
    v8f a0 = {}, a1 = {}, a2 = {}, a3 = {};

    // set 0 @ k, set 1 @ k+32
    v16bf xa  = load_fragA_bf16(Arow, 0);
    v16bf xb0 = load_fragB_bf16(B0, 0);
    v16bf xb1 = load_fragB_bf16(B1, 0);
    v16bf xb2 = load_fragB_bf16(B2, 0);
    v16bf xb3 = load_fragB_bf16(B3, 0);
    v16bf ya  = load_fragA_bf16(Arow, 32);
    v16bf yb0 = load_fragB_bf16(B0, 32);
    v16bf yb1 = load_fragB_bf16(B1, 32);
    v16bf yb2 = load_fragB_bf16(B2, 32);
    v16bf yb3 = load_fragB_bf16(B3, 32);

#pragma unroll 1
    for (int k = 0; k < NN - 64; k += 64) {
        // compute set 0 (k), then refill set 0 with k+64
        a0 = wmma_bf16(xa, xb0, a0);
        a1 = wmma_bf16(xa, xb1, a1);
        a2 = wmma_bf16(xa, xb2, a2);
        a3 = wmma_bf16(xa, xb3, a3);
        xa  = load_fragA_bf16(Arow, k + 64);
        xb0 = load_fragB_bf16(B0, k + 64);
        xb1 = load_fragB_bf16(B1, k + 64);
        xb2 = load_fragB_bf16(B2, k + 64);
        xb3 = load_fragB_bf16(B3, k + 64);
        // compute set 1 (k+32), then refill set 1 with k+96
        a0 = wmma_bf16(ya, yb0, a0);
        a1 = wmma_bf16(ya, yb1, a1);
        a2 = wmma_bf16(ya, yb2, a2);
        a3 = wmma_bf16(ya, yb3, a3);
        ya  = load_fragA_bf16(Arow, k + 96);
        yb0 = load_fragB_bf16(B0, k + 96);
        yb1 = load_fragB_bf16(B1, k + 96);
        yb2 = load_fragB_bf16(B2, k + 96);
        yb3 = load_fragB_bf16(B3, k + 96);
    }
    // epilogue: k = NN-64 (set 0), k = NN-32 (set 1)
    a0 = wmma_bf16(xa, xb0, a0);
    a1 = wmma_bf16(xa, xb1, a1);
    a2 = wmma_bf16(xa, xb2, a2);
    a3 = wmma_bf16(xa, xb3, a3);
    a0 = wmma_bf16(ya, yb0, a0);
    a1 = wmma_bf16(ya, yb1, a1);
    a2 = wmma_bf16(ya, yb2, a2);
    a3 = wmma_bf16(ya, yb3, a3);

    float* C = Xhat + (size_t)b * CC * MODES2;
    store_tile(C, MODES2, tm * 16, n0,      a0);
    store_tile(C, MODES2, tm * 16, n0 + 16, a1);
    store_tile(C, MODES2, tm * 16, n0 + 32, a2);
    store_tile(C, MODES2, tm * 16, n0 + 48, a3);
}

// ---- per-mode complex channel mixing (small: VALU) -------------------------
__global__ void mix_kernel(const float* __restrict__ Xhat,
                           const float* __restrict__ spec_w_l,
                           const float* __restrict__ freq_w_l,
                           float* __restrict__ Yhat) {
    int idx = blockIdx.x * blockDim.x + threadIdx.x;   // b*8192 + o*64 + m
    int m = idx & 63;
    int o = (idx >> 6) & 127;
    int b = idx >> 13;
    float fw = freq_w_l[m];
    const float* Xv = Xhat + (size_t)b * CC * MODES2;
    float yr = 0.0f, yi = 0.0f;
    for (int i = 0; i < CC; ++i) {
        float xr = Xv[i * MODES2 + 2 * m];
        float xi = Xv[i * MODES2 + 2 * m + 1];
        const float* w = spec_w_l + (((size_t)i * CC + o) * 64 + m) * 2;
        float wr = w[0], wi = w[1];
        yr += xr * wr - xi * wi;
        yi += xr * wi + xi * wr;
    }
    float* Yv = Yhat + (size_t)b * CC * MODES2;
    Yv[o * MODES2 + 2 * m]     = yr * fw;
    Yv[o * MODES2 + 2 * m + 1] = yi * fw;
}

// ---- inverse DFT GEMM: BufA(b)[o][n] = Yhat(b)[o][m2] * Gt[n][m2]^T --------
// Each wave: 16 o-rows x 64 n-cols; K=128 fully unrolled.
__global__ void __launch_bounds__(128, 1)
dft_inv_kernel(const float* __restrict__ Yhat,
               const bf16* __restrict__ Gt,
               float* __restrict__ BufA) {
    // grid (128, 2, B), block (32,4): tm = by*4+ty, n0 = bx*64
    int tm = blockIdx.y * 4 + threadIdx.y;
    int n0 = blockIdx.x * 64;
    int b  = blockIdx.z;
    const float* Arow = Yhat + ((size_t)b * CC + tm * 16 + (threadIdx.x & 15)) * MODES2;
    const bf16* B0 = Gt + (size_t)(n0 + (threadIdx.x & 15)) * MODES2;
    v8f a0 = {}, a1 = {}, a2 = {}, a3 = {};
#pragma unroll
    for (int k = 0; k < MODES2; k += 32) {
        v16bf a = load_fragA_f32(Arow, k);
        v16bf b0 = load_fragB_bf16(B0, k);
        v16bf b1 = load_fragB_bf16(B0 + (size_t)16 * MODES2, k);
        v16bf b2 = load_fragB_bf16(B0 + (size_t)32 * MODES2, k);
        v16bf b3 = load_fragB_bf16(B0 + (size_t)48 * MODES2, k);
        a0 = wmma_bf16(a, b0, a0);
        a1 = wmma_bf16(a, b1, a1);
        a2 = wmma_bf16(a, b2, a2);
        a3 = wmma_bf16(a, b3, a3);
    }
    float* C = BufA + (size_t)b * CC * NN;
    store_tile(C, NN, tm * 16, n0,      a0);
    store_tile(C, NN, tm * 16, n0 + 16, a1);
    store_tile(C, NN, tm * 16, n0 + 32, a2);
    store_tile(C, NN, tm * 16, n0 + 48, a3);
}

// ---- skip conv1x1 GEMM: async-to-LDS staging; accumulates into BufA --------
__global__ void skip_kernel(const bf16* __restrict__ Xb,
                            const float* __restrict__ skip_w_l,
                            float* __restrict__ BufA) {
    // grid (64, 8, B), block (32,8): block tile = 16 o-rows x 128 n-cols
    __shared__ bf16 sB[32][136];                       // K-panel x n, padded
    int lane = threadIdx.x & 31;
    int wave = threadIdx.y;                            // 0..7 -> n sub-tile
    int tm = blockIdx.y;
    int b  = blockIdx.z;
    int nbase0 = blockIdx.x * 128;
    const float* Arow = skip_w_l + (size_t)(tm * 16 + (lane & 15)) * CC;
    const bf16* Xv = Xb + (size_t)b * CC * NN;
    v8f acc = {};
    for (int k0 = 0; k0 < CC; k0 += 32) {
        __syncthreads();
        // stage K=32 x N=128 bf16 panel: 256 threads x 32B via async LDS loads
        int t  = wave * 32 + lane;
        int kk = t >> 3;
        int cc = (t & 7) * 16;
        const bf16* g = Xv + (size_t)(k0 + kk) * NN + nbase0 + cc;
        unsigned lds0 = (unsigned)(uintptr_t)&sB[kk][cc];
        asm volatile("global_load_async_to_lds_b128 %0, %1, off"
                     :: "v"(lds0), "v"((unsigned long long)g) : "memory");
        asm volatile("global_load_async_to_lds_b128 %0, %1, off"
                     :: "v"(lds0 + 16u), "v"((unsigned long long)(g + 8)) : "memory");
        asm volatile("s_wait_asynccnt 0x0" ::: "memory");
        __syncthreads();
        v16bf a = load_fragA_f32(Arow, k0);
        int half = lane >> 4;
        int n = wave * 16 + (lane & 15);
        v16bf bb;
#pragma unroll
        for (int e = 0; e < 16; ++e) bb[e] = sB[16 * half + e][n];
        acc = wmma_bf16(a, bb, acc);
    }
    int half = lane >> 4;
    int nn = nbase0 + wave * 16 + (lane & 15);
    float* C = BufA + (size_t)b * CC * NN;
#pragma unroll
    for (int v = 0; v < 8; ++v) {
        size_t i = (size_t)(tm * 16 + v + 8 * half) * NN + nn;
        C[i] += acc[v];                                 // add onto x_fno
    }
}

// ---- residual + bias + channel-LayerNorm + GELU ----------------------------
__global__ void combine_kernel(const float* __restrict__ BufA,
                               float* __restrict__ X, bf16* __restrict__ Xb,
                               const float* __restrict__ skip_b_l,
                               const float* __restrict__ ln_g_l,
                               const float* __restrict__ ln_b_l) {
    int idx = blockIdx.x * blockDim.x + threadIdx.x;   // B*N
    int n = idx & (NN - 1);
    int b = idx >> 13;
    const float* A = BufA + (size_t)b * CC * NN + n;
    float* Xp = X + (size_t)b * CC * NN + n;
    bf16*  Bp = Xb + (size_t)b * CC * NN + n;
    float mu = 0.0f, s2 = 0.0f;
    for (int c = 0; c < CC; ++c) {
        float v = A[(size_t)c * NN] + Xp[(size_t)c * NN] + skip_b_l[c];
        mu += v;
        s2 += v * v;
    }
    mu *= (1.0f / CC);
    float var = s2 * (1.0f / CC) - mu * mu;
    float rinv = rsqrtf(var + 1e-5f);
    for (int c = 0; c < CC; ++c) {
        float v = A[(size_t)c * NN] + Xp[(size_t)c * NN] + skip_b_l[c];
        float y = gelu_f((v - mu) * rinv * ln_g_l[c] + ln_b_l[c]);
        Xp[(size_t)c * NN] = y;
        Bp[(size_t)c * NN] = (bf16)y;
    }
}

// ---- projection: 128 -> 64 -> 32 -> 1 --------------------------------------
__global__ void proj_kernel(const float* __restrict__ X,
                            const float* __restrict__ p1w, const float* __restrict__ p1b,
                            const float* __restrict__ p2w, const float* __restrict__ p2b,
                            const float* __restrict__ p3w, const float* __restrict__ p3b,
                            float* __restrict__ out) {
    int idx = blockIdx.x * blockDim.x + threadIdx.x;   // B*N
    int n = idx & (NN - 1);
    int b = idx >> 13;
    const float* Xp = X + (size_t)b * CC * NN + n;
    float h1[64];
#pragma unroll
    for (int j = 0; j < 64; ++j) h1[j] = p1b[j];
    for (int c = 0; c < CC; ++c) {
        float xc = Xp[(size_t)c * NN];
#pragma unroll 8
        for (int j = 0; j < 64; ++j) h1[j] += p1w[j * CC + c] * xc;
    }
#pragma unroll
    for (int j = 0; j < 64; ++j) h1[j] = gelu_f(h1[j]);
    float h2[32];
    for (int j2 = 0; j2 < 32; ++j2) {
        float acc = p2b[j2];
#pragma unroll 8
        for (int j = 0; j < 64; ++j) acc += p2w[j2 * 64 + j] * h1[j];
        h2[j2] = gelu_f(acc);
    }
    float o = p3b[0];
#pragma unroll
    for (int j2 = 0; j2 < 32; ++j2) o += p3w[j2] * h2[j2];
    out[idx] = o;
}

extern "C" void kernel_launch(void* const* d_in, const int* in_sizes, int n_in,
                              void* d_out, int out_size, void* d_ws, size_t ws_size,
                              hipStream_t stream) {
    const float* source  = (const float*)d_in[0];
    const float* lift1_w = (const float*)d_in[1];
    const float* lift1_b = (const float*)d_in[2];
    const float* lift2_w = (const float*)d_in[3];
    const float* lift2_b = (const float*)d_in[4];
    const float* spec_w  = (const float*)d_in[5];
    const float* freq_w  = (const float*)d_in[6];
    const float* skip_w  = (const float*)d_in[7];
    const float* skip_b  = (const float*)d_in[8];
    const float* ln_g    = (const float*)d_in[9];
    const float* ln_b    = (const float*)d_in[10];
    const float* proj1_w = (const float*)d_in[11];
    const float* proj1_b = (const float*)d_in[12];
    const float* proj2_w = (const float*)d_in[13];
    const float* proj2_b = (const float*)d_in[14];
    const float* proj3_w = (const float*)d_in[15];
    const float* proj3_b = (const float*)d_in[16];

    char* ws = (char*)d_ws;
    size_t off = 0;
    float* X    = (float*)(ws + off); off += (size_t)BB * CC * NN * sizeof(float);
    float* BufA = (float*)(ws + off); off += (size_t)BB * CC * NN * sizeof(float);
    bf16*  Xb   = (bf16*)(ws + off);  off += (size_t)BB * CC * NN * sizeof(bf16);
    float* Xhat = (float*)(ws + off); off += (size_t)BB * CC * MODES2 * sizeof(float);
    float* Yhat = (float*)(ws + off); off += (size_t)BB * CC * MODES2 * sizeof(float);
    bf16*  Ft   = (bf16*)(ws + off);  off += (size_t)MODES2 * NN * sizeof(bf16);
    bf16*  Gt   = (bf16*)(ws + off);  off += (size_t)NN * MODES2 * sizeof(bf16);

    init_dft_kernel<<<(MODES2 * NN) / 256, 256, 0, stream>>>(Ft, Gt);
    lift_kernel<<<(BB * NN) / 256, 256, 0, stream>>>(source, lift1_w, lift1_b,
                                                     lift2_w, lift2_b, X, Xb);
    for (int l = 0; l < NLAYERS; ++l) {
        dft_fwd_kernel<<<dim3(2, 2, BB), dim3(32, 4), 0, stream>>>(Xb, Ft, Xhat);
        mix_kernel<<<(BB * NN) / 256, 256, 0, stream>>>(
            Xhat, spec_w + (size_t)l * CC * CC * 64 * 2, freq_w + l * 64, Yhat);
        dft_inv_kernel<<<dim3(128, 2, BB), dim3(32, 4), 0, stream>>>(Yhat, Gt, BufA);
        skip_kernel<<<dim3(64, 8, BB), dim3(32, 8), 0, stream>>>(
            Xb, skip_w + (size_t)l * CC * CC, BufA);
        combine_kernel<<<(BB * NN) / 256, 256, 0, stream>>>(
            BufA, X, Xb, skip_b + l * CC, ln_g + l * CC, ln_b + l * CC);
    }
    proj_kernel<<<(BB * NN) / 256, 256, 0, stream>>>(X, proj1_w, proj1_b, proj2_w,
                                                     proj2_b, proj3_w, proj3_b,
                                                     (float*)d_out);
}
